// DistinctivenessLoss_21749714387508
// MI455X (gfx1250) — matrix operand order repaired and checked by reading, not compiled, not run on hardware
//
#include <hip/hip_runtime.h>

typedef __attribute__((ext_vector_type(16))) _Float16 v16h;
typedef __attribute__((ext_vector_type(8)))  float    v8f;

#define BATCH 4
#define CH 48
#define HW 4096
#define W64 64

// ---------------- workspace layout ----------------
// P  : 3 sets (d1, d2all, q2) x B x 4096 descriptors x 96 f16 (48 hi + 48 lo)
// AA : 3 x B x 4096 f32  (squared norms of (h+l) vectors)
// S2 : B x 4096 f32      (gathered attention2 scores)
// MX : 2 x B x 4096 int  (row counts, z = b*2 + mat)
// LB : B f32 per-batch distinctiveness loss
// GP : 2*B f32 per-image peaky gap sums
static constexpr size_t P_BYTES  = 3ull * BATCH * HW * 96 * 2;   // 9,437,184
static constexpr size_t AA_OFF   = P_BYTES;
static constexpr size_t AA_BYTES = 3ull * BATCH * HW * 4;        // 196,608
static constexpr size_t S2_OFF   = AA_OFF + AA_BYTES;
static constexpr size_t S2_BYTES = (size_t)BATCH * HW * 4;       // 65,536
static constexpr size_t MX_OFF   = S2_OFF + S2_BYTES;
static constexpr size_t MX_BYTES = 2ull * BATCH * HW * 4;        // 131,072
static constexpr size_t LB_OFF   = MX_OFF + MX_BYTES;
static constexpr size_t GP_OFF   = LB_OFF + 64;

// ------------- kernel 1: normalize, hi/lo split, gather -------------
__global__ __launch_bounds__(256)
void dl_prep_kernel(const float* __restrict__ x1, const float* __restrict__ x2,
                    const float* __restrict__ a2, const int* __restrict__ pos,
                    _Float16* __restrict__ P, float* __restrict__ AA,
                    float* __restrict__ S2) {
  int blk = blockIdx.x;                  // BATCH*3*16 blocks
  int j   = (blk & 15) * 256 + threadIdx.x;
  int set = (blk >> 4) % 3;              // 0: d1, 1: d2all, 2: q2
  int b   = blk / 48;

  const float* src = (set == 0) ? (x1 + (size_t)b * CH * HW)
                                : (x2 + (size_t)b * CH * HW);
  int p = j;
  if (set == 2) {
    int r = pos[((size_t)b * 2) * HW + j];
    int c = pos[((size_t)b * 2 + 1) * HW + j];
    p = r * W64 + c;
    S2[(size_t)b * HW + j] = a2[(size_t)b * HW + p];
  }

  float v[CH];
  float ss = 0.0f;
#pragma unroll
  for (int c = 0; c < CH; ++c) {
    v[c] = src[(size_t)c * HW + p];
    ss += v[c] * v[c];
  }
  float inv = 1.0f / fmaxf(sqrtf(ss), 1e-12f);

  _Float16* row = P + ((size_t)(set * BATCH + b) * HW + j) * 96;
  float aa = 0.0f;
#pragma unroll
  for (int c = 0; c < CH; ++c) {
    float x = v[c] * inv;
    _Float16 h = (_Float16)x;
    float    hf = (float)h;
    _Float16 l = (_Float16)(x - hf);
    row[c]      = h;
    row[48 + c] = l;
    float u = hf + (float)l;
    aa += u * u;
  }
  AA[(size_t)(set * BATCH + b) * HW + j] = aa;
}

// ------------- kernel 2: WMMA dot + threshold count -------------
// grid: x = 32 (row blocks of 128), y = 8 (b*2 + mat)
// mat 0: A = d1   (set 0), B = d2all (set 1)   -> mx1
// mat 1: A = q2   (set 2), B = d1    (set 0)   -> mx2
//
// Concatenated operand of length 192 over the compact 96-f16 rows:
//   A content: [h, l, h, l]  -> elem(kp) = ((kp/48)&1)*48 + kp%48
//   B content: [h, h, l, l]  -> elem(kp) = ((kp/48)>>1)*48 + kp%48
// For fixed (step, laneGroup), each fragment half/whole is a run of
// consecutive dwords that never crosses a 48-element region boundary and is
// 16B aligned -> fragments load as global b128 pairs, no gathers, no LDS.
__global__ __launch_bounds__(256)
void dl_count_kernel(const _Float16* __restrict__ P, const float* __restrict__ AA,
                     int* __restrict__ MX) {
  int z = blockIdx.y;
  int b = z >> 1, mat = z & 1;
  int setA = mat ? 2 : 0;
  int setB = mat ? 0 : 1;
  const _Float16* Pa = P + (size_t)(setA * BATCH + b) * HW * 96;
  const _Float16* Pb = P + (size_t)(setB * BATCH + b) * HW * 96;
  const float* AAa = AA + (size_t)(setA * BATCH + b) * HW;
  const float* BBb = AA + (size_t)(setB * BATCH + b) * HW;

  int tid  = threadIdx.x;
  int lane = tid & 31, wave = tid >> 5;
  int g = lane >> 4, ln = lane & 15;
  int rowbase = blockIdx.x * 128 + wave * 16;

  // ---- A fragments for all 6 K-steps, kept in registers for the whole loop.
  // A layout: row M = lane&15; VGPR p holds K pair at
  //   kp = 32*s + laneGroup*16 + 2*p
  v16h aF[6];
  {
    const uint4* ArV = (const uint4*)(Pa + (size_t)(rowbase + ln) * 96); // 12/row
#pragma unroll
    for (int s = 0; s < 6; ++s) {
      union { v16h v; uint4 q[2]; } t;
#pragma unroll
      for (int h = 0; h < 2; ++h) {
        int k0 = 32 * s + 16 * g + h * 8;                // start kp, mult of 8
        int e0 = ((k0 / 48) & 1) * 48 + (k0 % 48);       // f16 elem, mult of 8
        t.q[h] = ArV[e0 >> 3];
      }
      aF[s] = t.v;
    }
  }

  // B fragment dword-group offsets per step (uint4 index into a 12-uint4 row):
  //   kp = 32*s + laneGroup*16 + 2*p  -> 16 consecutive f16 = 2 uint4
  int eB[6];
#pragma unroll
  for (int s = 0; s < 6; ++s) {
    int k0 = 32 * s + 16 * g;                            // mult of 16
    int e0 = ((k0 / 48) >> 1) * 48 + (k0 % 48);          // mult of 16
    eB[s] = e0 >> 3;
  }

  // D-tile row ownership: row = rowbase + g*8 + r
  float aaR[8];
#pragma unroll
  for (int r = 0; r < 8; ++r) aaR[r] = AAa[rowbase + g * 8 + r];

  int cnt[8] = {0, 0, 0, 0, 0, 0, 0, 0};

  for (int jt = 0; jt < 256; ++jt) {                     // 256 column tiles
    int col = jt * 16 + ln;                              // lane's output column
    const uint4* rB = (const uint4*)(Pb + (size_t)col * 96);
    float bbn = BBb[col];
    v8f acc = {};
#pragma unroll
    for (int s = 0; s < 6; ++s) {
      union { v16h v; uint4 q[2]; } t;
      t.q[0] = rB[eB[s]];
      t.q[1] = rB[eB[s] + 1];
      acc = __builtin_amdgcn_wmma_f32_16x16x32_f16(
          false, aF[s], false, t.v, (short)0, acc, false, false);
    }
#pragma unroll
    for (int r = 0; r < 8; ++r) {
      float sq = aaR[r] + bbn - 2.0f * acc[r];
      cnt[r] += (sq < 1.0f) ? 1 : 0;
    }
  }

  // reduce counts across the 16 lanes sharing each row (masks<16 stay in-half)
#pragma unroll
  for (int r = 0; r < 8; ++r) {
    int v = cnt[r];
    v += __shfl_xor(v, 1);
    v += __shfl_xor(v, 2);
    v += __shfl_xor(v, 4);
    v += __shfl_xor(v, 8);
    cnt[r] = v;
  }
  if (ln == 0) {
#pragma unroll
    for (int r = 0; r < 8; ++r)
      MX[(size_t)z * HW + rowbase + g * 8 + r] = cnt[r];
  }
}

// ------------- kernel 3: per-batch L1 distinctiveness loss -------------
__global__ __launch_bounds__(256)
void dl_loss_kernel(const float* __restrict__ a1, const float* __restrict__ S2,
                    const int* __restrict__ MX, float* __restrict__ LB) {
  int b = blockIdx.x;
  __shared__ float red[256];
  float s = 0.0f;
  for (int i = threadIdx.x; i < HW; i += 256) {
    float m1 = (float)MX[(size_t)(2 * b) * HW + i];
    float t1 = 1.0f / sqrtf(sqrtf(1.0f + m1));          // (1+m)^-0.25
    s += fabsf(a1[(size_t)b * HW + i] - t1);
    float m2 = (float)MX[(size_t)(2 * b + 1) * HW + i];
    float t2 = 1.0f / sqrtf(sqrtf(1.0f + m2));
    s += fabsf(S2[(size_t)b * HW + i] - t2);
  }
  red[threadIdx.x] = s;
  __syncthreads();
  for (int k = 128; k > 0; k >>= 1) {
    if (threadIdx.x < k) red[threadIdx.x] += red[threadIdx.x + k];
    __syncthreads();
  }
  if (threadIdx.x == 0) LB[b] = red[0] * (1.0f / (float)HW);
}

// ------------- kernel 4: peaky gap sums (separable pooling) -------------
__global__ __launch_bounds__(256)
void dl_peaky_kernel(const float* __restrict__ a1, const float* __restrict__ a2,
                     float* __restrict__ GP) {
  int blk = blockIdx.x;                 // 0..7 : tensor*4 + batch
  int t = blk >> 2, b = blk & 3;
  const float* a = (t == 0 ? a1 : a2) + (size_t)b * HW;

  __shared__ float sA[HW];
  __shared__ float sS[HW];
  __shared__ float sT[HW];
  __shared__ float red[256];

  for (int i = threadIdx.x; i < HW; i += 256) sA[i] = a[i];
  __syncthreads();

  // sali = 3x3 avg pool, zero pad, /9 (count includes padding)
  for (int i = threadIdx.x; i < HW; i += 256) {
    int r = i >> 6, c = i & 63;
    float s = 0.0f;
    for (int dr = -1; dr <= 1; ++dr)
      for (int dc = -1; dc <= 1; ++dc) {
        int rr = r + dr, cc = c + dc;
        if (rr >= 0 && rr < W64 && cc >= 0 && cc < W64) s += sA[rr * W64 + cc];
      }
    sS[i] = s * (1.0f / 9.0f);
  }
  __syncthreads();

  // horizontal 33-tap pass: max -> sA, sum -> sT
  for (int i = threadIdx.x; i < HW; i += 256) {
    int r = i >> 6, c = i & 63;
    int c0 = c - 16 > 0 ? c - 16 : 0;
    int c1 = c + 16 < 63 ? c + 16 : 63;
    float mx = -3.0e38f, sm = 0.0f;
    for (int cc = c0; cc <= c1; ++cc) {
      float v = sS[r * W64 + cc];
      mx = fmaxf(mx, v);
      sm += v;
    }
    sA[i] = mx;
    sT[i] = sm;
  }
  __syncthreads();

  // vertical 33-tap pass + gap accumulation
  float part = 0.0f;
  for (int i = threadIdx.x; i < HW; i += 256) {
    int r = i >> 6, c = i & 63;
    int r0 = r - 16 > 0 ? r - 16 : 0;
    int r1 = r + 16 < 63 ? r + 16 : 63;
    float mx = -3.0e38f, sm = 0.0f;
    for (int rr = r0; rr <= r1; ++rr) {
      mx = fmaxf(mx, sA[rr * W64 + c]);
      sm += sT[rr * W64 + c];
    }
    part += mx - sm * (1.0f / 1089.0f);   // max_pool - avg_pool (÷(33*33))
  }
  red[threadIdx.x] = part;
  __syncthreads();
  for (int k = 128; k > 0; k >>= 1) {
    if (threadIdx.x < k) red[threadIdx.x] += red[threadIdx.x + k];
    __syncthreads();
  }
  if (threadIdx.x == 0) GP[blk] = red[0];
}

// ------------- kernel 5: final scalar combine -------------
__global__ void dl_final_kernel(const float* __restrict__ LB,
                                const float* __restrict__ GP,
                                float* __restrict__ out) {
  if (threadIdx.x == 0) {
    float l = 0.0f;
    for (int b = 0; b < BATCH; ++b) l += LB[b];
    l *= 0.25f;
    float g1 = 0.0f, g2 = 0.0f;
    for (int b = 0; b < BATCH; ++b) { g1 += GP[b]; g2 += GP[4 + b]; }
    float p1 = fmaxf(1.0f - g1 * (1.0f / 16384.0f), 0.0f);
    float p2 = fmaxf(1.0f - g2 * (1.0f / 16384.0f), 0.0f);
    out[0] = l + 0.1f * (p1 + p2);       // 0.2 * (p1+p2)/2
  }
}

extern "C" void kernel_launch(void* const* d_in, const int* in_sizes, int n_in,
                              void* d_out, int out_size, void* d_ws, size_t ws_size,
                              hipStream_t stream) {
  const float* x1  = (const float*)d_in[0];
  const float* x2  = (const float*)d_in[1];
  const float* a1  = (const float*)d_in[2];
  const float* a2  = (const float*)d_in[3];
  const int*   pos = (const int*)d_in[4];
  float* out = (float*)d_out;

  char* w = (char*)d_ws;
  _Float16* P  = (_Float16*)w;
  float*    AA = (float*)(w + AA_OFF);
  float*    S2 = (float*)(w + S2_OFF);
  int*      MX = (int*)(w + MX_OFF);
  float*    LB = (float*)(w + LB_OFF);
  float*    GP = (float*)(w + GP_OFF);

  dl_prep_kernel<<<dim3(BATCH * 3 * 16), 256, 0, stream>>>(x1, x2, a2, pos, P, AA, S2);
  dl_count_kernel<<<dim3(32, 2 * BATCH), 256, 0, stream>>>(P, AA, MX);
  dl_loss_kernel<<<dim3(BATCH), 256, 0, stream>>>(a1, S2, MX, LB);
  dl_peaky_kernel<<<dim3(2 * BATCH), 256, 0, stream>>>(a1, a2, GP);
  dl_final_kernel<<<dim3(1), 32, 0, stream>>>(LB, GP, out);
}
